// Attention_61856118997337
// MI455X (gfx1250) — matrix operand order repaired and checked by reading, not compiled, not run on hardware
//
#include <hip/hip_runtime.h>
#include <hip/hip_bf16.h>

typedef __attribute__((ext_vector_type(16))) __bf16 v16bf;
typedef __attribute__((ext_vector_type(8)))  __bf16 v8bf;
typedef __attribute__((ext_vector_type(8)))  float  v8f;

#define DIMD   1024
#define SEQ    2048
#define NBATCH 32

#if defined(__has_builtin)
#if __has_builtin(__builtin_amdgcn_sched_barrier)
#define SCHED_FENCE() __builtin_amdgcn_sched_barrier(0)
#endif
#endif
#ifndef SCHED_FENCE
#define SCHED_FENCE() asm volatile("" ::: "memory")
#endif

__device__ __forceinline__ float fast_tanh(float x) {
    // tanh(x) = 1 - 2/(exp(2x)+1); exact saturation at +-inf
    const float e2 = __expf(2.f * x);
    return 1.f - __fdividef(2.f, e2 + 1.f);
}

// ---------------- hid_proj = hidden @ W1_hid + b1  (tiny GEMV) ----------------
__global__ void k_hidproj(const float* __restrict__ hidden,
                          const float* __restrict__ W1,
                          const float* __restrict__ b1,
                          float* __restrict__ hp) {
    const int e = blockIdx.x * 256 + threadIdx.x;
    const int b = blockIdx.y;
    const float* Wh   = W1 + (size_t)DIMD * DIMD;      // W1[D:, :]
    const float* hrow = hidden + (size_t)b * DIMD;
    float acc = 0.f;
    #pragma unroll 4
    for (int d = 0; d < DIMD; ++d)
        acc += hrow[d] * Wh[(size_t)d * DIMD + e];
    hp[(size_t)b * DIMD + e] = acc + b1[e];
}

// -------- W1_enc (row-major fp32, [k][n]) -> Wt (column-major bf16, [n][k]) ---
__global__ void k_w1t(const float* __restrict__ W1, __bf16* __restrict__ Wt) {
    const int idx = blockIdx.x * 256 + threadIdx.x;   // coalesced read
    const int n = idx & (DIMD - 1);
    const int k = idx >> 10;
    Wt[(size_t)n * DIMD + k] = (__bf16)W1[(size_t)k * DIMD + n];
}

// ------- fused: e[b,s] = w2 . tanh(enc@W1_enc + hp)  via bf16 WMMA ------------
__launch_bounds__(256, 1)
__global__ void k_score(const float* __restrict__ enc,
                        const __bf16* __restrict__ Wt,
                        const float* __restrict__ hp,
                        const float* __restrict__ w2,
                        float* __restrict__ e_out) {
    __shared__ __align__(16) __bf16 Asm[2][32 * 32];   // 2 x (32 rows x 32 k)
    __shared__ float e_sm[32];

    const int tid      = threadIdx.x;
    const int wave     = tid >> 5;
    const int lane     = tid & 31;
    const int laneN    = lane & 15;
    const int laneHalf = lane >> 4;

    const int r0 = blockIdx.x * 32;        // flat row = b*SEQ + s ; slab in one batch
    const int b  = r0 >> 11;
    const int s0 = r0 & (SEQ - 1);

    if (tid < 32) e_sm[tid] = 0.f;

    // cooperative A staging: each thread converts 4 fp32 -> 4 bf16
    const int sRow = tid >> 3;
    const int sCol = (tid & 7) * 4;
    const float* aBase = enc + (size_t)(r0 + sRow) * DIMD + sCol;
    {
        const float4 v = *(const float4*)aBase;
        __bf16* dst = &Asm[0][sRow * 32 + sCol];
        dst[0] = (__bf16)v.x; dst[1] = (__bf16)v.y;
        dst[2] = (__bf16)v.z; dst[3] = (__bf16)v.w;
    }
    __syncthreads();

    v8f acc[2][8];
    #pragma unroll
    for (int mm = 0; mm < 2; ++mm)
        #pragma unroll
        for (int t = 0; t < 8; ++t) { v8f z = {}; acc[mm][t] = z; }

    const int nBase = wave * 128;
    const __bf16* wBase = Wt + (size_t)(nBase + laneN) * DIMD + laneHalf * 16;

    for (int kt = 0; kt < 32; ++kt) {
        const int slot = kt & 1;

        // --- issue region: A-prefetch load + 16 B loads + 4 A ds loads -------
        float4 av;
        if (kt + 1 < 32)
            av = *(const float4*)(aBase + (kt + 1) * 32);   // next A tile (global)

        v16bf bfrag[8];                                     // 16 global b128 loads
        #pragma unroll
        for (int t = 0; t < 8; ++t)
            bfrag[t] = *(const v16bf*)(wBase + (size_t)(t * 16) * DIMD + kt * 32);

        // A fragments (contiguous k, matching B): lanes 0-15: K 0-15,
        // lanes 16-31: K 16-31, row = lane&15
        v16bf afrag[2];
        #pragma unroll
        for (int mm = 0; mm < 2; ++mm) {
            const __bf16* ap = &Asm[slot][(mm * 16 + laneN) * 32];
            v8bf c0 = *(const v8bf*)(ap + laneHalf * 16);
            v8bf c1 = *(const v8bf*)(ap + laneHalf * 16 + 8);
            #pragma unroll
            for (int i = 0; i < 8; ++i) { afrag[mm][i] = c0[i]; afrag[mm][i + 8] = c1[i]; }
        }

        SCHED_FENCE();   // keep all loads above; all bfrags live => distinct VGPRs

        // --- compute region: 16 WMMAs drain the load clause ------------------
        #pragma unroll
        for (int t = 0; t < 8; ++t) {
            acc[0][t] = __builtin_amdgcn_wmma_f32_16x16x32_bf16(
                false, afrag[0], false, bfrag[t], (short)0, acc[0][t], false, false);
            acc[1][t] = __builtin_amdgcn_wmma_f32_16x16x32_bf16(
                false, afrag[1], false, bfrag[t], (short)0, acc[1][t], false, false);
        }

        // convert + store next A tile, interleaved with WMMAs by the scheduler
        if (kt + 1 < 32) {
            __bf16* dst = &Asm[slot ^ 1][sRow * 32 + sCol];
            dst[0] = (__bf16)av.x; dst[1] = (__bf16)av.y;
            dst[2] = (__bf16)av.z; dst[3] = (__bf16)av.w;
        }
        __syncthreads();
    }

    // epilogue: tanh + dot with w2 (C layout: VGPR i -> row i + 8*laneHalf, col = laneN)
    float eloc[2][8];
    #pragma unroll
    for (int mm = 0; mm < 2; ++mm)
        #pragma unroll
        for (int i = 0; i < 8; ++i) eloc[mm][i] = 0.f;

    #pragma unroll
    for (int t = 0; t < 8; ++t) {
        const int n = nBase + t * 16 + laneN;
        const float hpn = hp[(size_t)b * DIMD + n];
        const float w2n = w2[n];
        #pragma unroll
        for (int mm = 0; mm < 2; ++mm)
            #pragma unroll
            for (int i = 0; i < 8; ++i)
                eloc[mm][i] += fast_tanh(acc[mm][t][i] + hpn) * w2n;
    }

    // reduce across the 16 N-lanes of each half-wave
    #pragma unroll
    for (int mm = 0; mm < 2; ++mm)
        #pragma unroll
        for (int i = 0; i < 8; ++i) {
            float v = eloc[mm][i];
            v += __shfl_xor(v, 1, 32);
            v += __shfl_xor(v, 2, 32);
            v += __shfl_xor(v, 4, 32);
            v += __shfl_xor(v, 8, 32);
            eloc[mm][i] = v;
        }

    if (laneN == 0) {
        #pragma unroll
        for (int mm = 0; mm < 2; ++mm)
            #pragma unroll
            for (int i = 0; i < 8; ++i)
                atomicAdd(&e_sm[mm * 16 + i + 8 * laneHalf], eloc[mm][i]);
    }
    __syncthreads();
    if (tid < 32) e_out[(size_t)b * SEQ + s0 + tid] = e_sm[tid];
}

// ---------------- softmax over S per batch (also zero d_out) ------------------
__global__ void k_softmax(float* __restrict__ e, float* __restrict__ out) {
    const int b = blockIdx.x;
    const int tid = threadIdx.x;
    __shared__ float red[256];
    float* eb = e + (size_t)b * SEQ;

    float m = -3.4e38f;
    for (int s = tid; s < SEQ; s += 256) m = fmaxf(m, eb[s]);
    red[tid] = m; __syncthreads();
    for (int off = 128; off > 0; off >>= 1) {
        if (tid < off) red[tid] = fmaxf(red[tid], red[tid + off]);
        __syncthreads();
    }
    m = red[0]; __syncthreads();

    float sum = 0.f;
    for (int s = tid; s < SEQ; s += 256) {
        const float ex = __expf(eb[s] - m);
        eb[s] = ex;
        sum += ex;
    }
    red[tid] = sum; __syncthreads();
    for (int off = 128; off > 0; off >>= 1) {
        if (tid < off) red[tid] += red[tid + off];
        __syncthreads();
    }
    const float inv = 1.f / red[0];
    for (int s = tid; s < SEQ; s += 256) eb[s] *= inv;

    #pragma unroll
    for (int j = 0; j < 4; ++j) out[(size_t)b * DIMD + j * 256 + tid] = 0.f;
}

// ---------------- context[b,d] = sum_s alpha[b,s] * enc[b,s,d] ----------------
__global__ void k_context(const float* __restrict__ alpha,
                          const float* __restrict__ enc,
                          float* __restrict__ out) {
    const int d = blockIdx.y * 256 + threadIdx.x;
    const int b = blockIdx.z;
    const int sBase = blockIdx.x * 256;
    const float* ab = alpha + (size_t)b * SEQ + sBase;
    const float* eb = enc + ((size_t)b * SEQ + sBase) * DIMD + d;
    float acc = 0.f;
    #pragma unroll 4
    for (int s = 0; s < 256; ++s)
        acc += ab[s] * eb[(size_t)s * DIMD];
    atomicAdd(&out[(size_t)b * DIMD + d], acc);
}

extern "C" void kernel_launch(void* const* d_in, const int* in_sizes, int n_in,
                              void* d_out, int out_size, void* d_ws, size_t ws_size,
                              hipStream_t stream) {
    const float* hidden = (const float*)d_in[0];
    const float* enc    = (const float*)d_in[1];
    const float* W1     = (const float*)d_in[2];
    const float* b1     = (const float*)d_in[3];
    const float* w2     = (const float*)d_in[4];
    float* out = (float*)d_out;

    char* ws = (char*)d_ws;
    __bf16* Wt = (__bf16*)ws;                                              // 2 MB
    float*  hp = (float*)(ws + (size_t)DIMD * DIMD * sizeof(__bf16));      // 128 KB
    float*  e  = (float*)(ws + (size_t)DIMD * DIMD * sizeof(__bf16)
                             + (size_t)NBATCH * DIMD * sizeof(float));     // 256 KB

    k_w1t    <<<dim3(DIMD * DIMD / 256), 256, 0, stream>>>(W1, Wt);
    k_hidproj<<<dim3(DIMD / 256, NBATCH), 256, 0, stream>>>(hidden, W1, b1, hp);
    k_score  <<<dim3(NBATCH * SEQ / 32), 256, 0, stream>>>(enc, Wt, hp, w2, e);
    k_softmax<<<dim3(NBATCH), 256, 0, stream>>>(e, out);
    k_context<<<dim3(SEQ / 256, DIMD / 256, NBATCH), 256, 0, stream>>>(e, enc, out);
}
